// BilinearPooling_47313359733154
// MI455X (gfx1250) — matrix operand order repaired and checked by reading, not compiled, not run on hardware
//
#include <hip/hip_runtime.h>
#include <hip/hip_bf16.h>

typedef _Float16 half16 __attribute__((ext_vector_type(16)));
typedef float    float8 __attribute__((ext_vector_type(8)));

static constexpr int Bb = 16, Nn = 64, Vv = 16, Dd = 128, Ee = 128, Oo = 256;
// W viewed as [R, E] with R = O*D = 32768 (row r = o*D + d, contiguous in e)
static constexpr int Rr = Oo * Dd;          // 32768
static constexpr int R_SHIFT = 15;          // Rr == 1 << 15

__device__ __forceinline__ float8 wmma_f16(half16 a, half16 b, float8 c) {
    // v_wmma_f32_16x16x32_f16  (8-arg codegen-confirmed form)
    return __builtin_amdgcn_wmma_f32_16x16x32_f16(false, a, false, b,
                                                  (short)0, c, false, false);
}

// A-fragment (16x32, f16) from row-major f32 [16 x lda], K window [k0, k0+32).
// Layout used (contiguous k): lane L -> row L&15; halves j=0..15 -> K=j (+16 if L>=16).
__device__ __forceinline__ half16 load_a_f32(const float* __restrict__ A, int lda,
                                             int k0, int lane) {
    const int m   = lane & 15;
    const int hi8 = (lane >> 4) << 3;
    const float* row = A + m * lda + k0;
    half16 a;
#pragma unroll
    for (int j = 0; j < 16; ++j) {
        int k = j + (hi8 << 1);               // {0..15} (+16 for hi lanes)
        a[j] = (_Float16)row[k];
    }
    return a;
}

// B-fragment (32x16) from Bt row-major [16 x ldb] f32 (Bt[n,k] == B[k,n]).
// Layout used (contiguous k): lane L -> column L&15; halves j -> K = j + (L>=16 ? 16 : 0).
__device__ __forceinline__ half16 load_bt_f32(const float* __restrict__ Bt, int ldb,
                                              int k0, int lane) {
    const int n    = lane & 15;
    const int koff = (lane >> 4) << 4;
    const float* row = Bt + n * ldb + k0 + koff;
    half16 b;
#pragma unroll
    for (int j = 0; j < 16; ++j) b[j] = (_Float16)row[j];
    return b;
}

// Same B-fragment from an f16 source: one aligned 32B vector load per lane.
__device__ __forceinline__ half16 load_bt_f16(const _Float16* __restrict__ Bt, int ldb,
                                              int k0, int lane) {
    const int n    = lane & 15;
    const int koff = (lane >> 4) << 4;
    return *(const half16*)(Bt + n * ldb + k0 + koff);
}

// -------- GEMM1: T[b,v,r] = sum_e veh[b,v,e] * W[r,e]   (T stored f16) --------
// grid: 512 r-groups x 16 batches (batch fastest -> concurrent blocks share W
// tile rows in L2). block: 128 threads = 4 waves, wave = one 16x16 tile, K=128.
__global__ __launch_bounds__(128) void bp_gemm1(const float* __restrict__ veh,
                                                const float* __restrict__ W,
                                                _Float16* __restrict__ T) {
    const int lane   = threadIdx.x & 31;
    const int wave   = threadIdx.x >> 5;
    const int b      = blockIdx.x & 15;
    const int rgroup = blockIdx.x >> 4;
    const int r0     = (rgroup * 4 + wave) << 4;

    const float* A = veh + (size_t)b * (Vv * Ee);   // [16 x 128]
    half16 af[4];
#pragma unroll
    for (int kc = 0; kc < 4; ++kc) af[kc] = load_a_f32(A, Ee, kc * 32, lane);

    const float* Bt = W + (size_t)r0 * Ee;          // rows r0..r0+15 of [R x E]
    float8 acc = {};
#pragma unroll
    for (int kc = 0; kc < 4; ++kc) {
        half16 bf = load_bt_f32(Bt, Ee, kc * 32, lane);
        acc = wmma_f16(af[kc], bf, acc);
    }

    // C layout: VGPR i -> row (v) = i + 8*hi, col = r0 + (lane&15)
    const int n  = lane & 15;
    const int hi = lane >> 4;
    _Float16* Tb = T + (((size_t)(b * Vv)) << R_SHIFT) + r0 + n;
#pragma unroll
    for (int i = 0; i < 8; ++i) {
        int v = i + (hi << 3);
        Tb[(size_t)v << R_SHIFT] = (_Float16)acc[i];
    }
}

// -------- GEMM2: out[b,n,v,o] = sum_d node[b,n,d] * T[b,v,o*D+d] + bias[o] ----
// grid: 256 (b,v) x 4 o-groups. block: 128 = 4 waves; wave = one 16-row m-tile
// across four 16-col o-tiles (A fragments reused 4x). 16 WMMAs per wave.
__global__ __launch_bounds__(128) void bp_gemm2(const float* __restrict__ node,
                                                const _Float16* __restrict__ T,
                                                const float* __restrict__ bias,
                                                float* __restrict__ out) {
    const int lane = threadIdx.x & 31;
    const int wave = threadIdx.x >> 5;
    const int og   = blockIdx.x & 3;
    const int bv   = blockIdx.x >> 2;
    const int b    = bv >> 4;
    const int v    = bv & 15;
    const int m0   = wave << 4;

    const float* A = node + (size_t)(b * Nn + m0) * Dd;   // rows m0..m0+15
    half16 af[4];
#pragma unroll
    for (int kc = 0; kc < 4; ++kc) af[kc] = load_a_f32(A, Dd, kc * 32, lane);

    const _Float16* Tb = T + ((size_t)bv << R_SHIFT);     // Bt2[o,d] = Tb[o*128+d]
    const int n  = lane & 15;
    const int hi = lane >> 4;

#pragma unroll
    for (int j = 0; j < 4; ++j) {
        const int o0 = og * 64 + j * 16;
        const _Float16* Brow = Tb + (size_t)o0 * Dd;
        float8 acc = {};
#pragma unroll
        for (int kc = 0; kc < 4; ++kc) {
            half16 bf = load_bt_f16(Brow, Dd, kc * 32, lane);
            acc = wmma_f16(af[kc], bf, acc);
        }
        const int   o  = o0 + n;
        const float bo = bias[o];
#pragma unroll
        for (int i = 0; i < 8; ++i) {
            int nn = m0 + i + (hi << 3);
            // ((b*N + nn)*V + v)*O + o ; V*O = 4096, O = 256
            out[(((size_t)(b * Nn + nn) * Vv + v) << 8) + o] = acc[i] + bo;
        }
    }
}

extern "C" void kernel_launch(void* const* d_in, const int* in_sizes, int n_in,
                              void* d_out, int out_size, void* d_ws, size_t ws_size,
                              hipStream_t stream) {
    const float* node = (const float*)d_in[0];   // [16,64,128]
    const float* veh  = (const float*)d_in[1];   // [16,16,128]
    const float* W    = (const float*)d_in[2];   // [256,16384] == [32768,128]
    const float* bias = (const float*)d_in[3];   // [256]
    float* out = (float*)d_out;                  // [16,64,16,256]

    _Float16* T = (_Float16*)d_ws;               // 16 MiB: [B,V,R] f16

    // GEMM1: 512 r-groups * 16 batches, 4 waves per block
    bp_gemm1<<<dim3((Rr / 64) * Bb), dim3(128), 0, stream>>>(veh, W, T);
    // GEMM2: 256 (b,v) * 4 o-groups, 4 waves per block
    bp_gemm2<<<dim3(Bb * Vv * 4), dim3(128), 0, stream>>>(node, T, bias, out);
}
